// VGCNBlock_80693845557441
// MI455X (gfx1250) — matrix operand order repaired and checked by reading, not compiled, not run on hardware
//
#include <hip/hip_runtime.h>
#include <math.h>
#include <stdint.h>

typedef __attribute__((ext_vector_type(16))) _Float16 v16h;
typedef __attribute__((ext_vector_type(8)))  float    v8f;

#define NN 50000
#define NE 640000
#define DF 128
#define ALPHA_C 0.1f
#define EPS_C 0.005f
#define KDROP (NE / 2)

// ---------------------------------------------------------------------------
// Edge attention via WMMA: one wave32 per 16-edge tile.
// Computes Gram(dif_tile) with v_wmma_f32_16x16x32_f16, takes the diagonal
// as l2^2, then att = 1/(2*l2 + eps).
// Lane-variant K ranges are folded into load ADDRESSES (free) instead of
// register indices (which previously compiled to cndmask chains).
// ---------------------------------------------------------------------------
__global__ __launch_bounds__(256) void att_kernel(const float* __restrict__ feat,
                                                  const int* __restrict__ src,
                                                  const int* __restrict__ dst,
                                                  float* __restrict__ att)
{
    const int wave = threadIdx.x >> 5;
    const int lane = threadIdx.x & 31;
    const int tile = blockIdx.x * 8 + wave;   // NE/16 tiles, grid exact
    const int row  = lane & 15;               // edge within tile
    const int hf   = lane >> 4;               // which K-half this lane owns
    const int e    = tile * 16 + row;

    const int s = src[e];
    const int d = dst[e];
    const float4* F  = (const float4*)feat;
    const size_t  sb = (size_t)s * (DF / 4);  // row base in float4 units
    const size_t  db = (size_t)d * (DF / 4);

    // float4-granular offsets within a row (lane-dependent, loop-invariant)
    const int offA0 = 2 * hf;       // floats kb + 8*hf      -> float4 idx +2*hf
    const int offA1 = 4 + 2 * hf;   // floats kb + 16 + 8*hf
    const int offB  = 2 * hf;       // floats kb + 8*hf (+16 for q>=2)

    v8f acc = {};
    for (int c = 0; c < 4; ++c) {             // 4 chunks of K=32 -> K=128
        const int kb4 = c * 8;                // chunk base in float4 units
        v16h a, b;
        // A (16x32 f16): lane<16 holds K 0..7 & 16..23 of row M=lane,
        //                lane>=16 holds K 8..15 & 24..31 of row M=lane-16.
        #pragma unroll
        for (int q = 0; q < 2; ++q) {
            float4 x = F[sb + kb4 + offA0 + q];
            float4 y = F[db + kb4 + offA0 + q];
            a[q * 4 + 0] = (_Float16)(x.x - y.x);
            a[q * 4 + 1] = (_Float16)(x.y - y.y);
            a[q * 4 + 2] = (_Float16)(x.z - y.z);
            a[q * 4 + 3] = (_Float16)(x.w - y.w);
        }
        #pragma unroll
        for (int q = 0; q < 2; ++q) {
            float4 x = F[sb + kb4 + offA1 + q];
            float4 y = F[db + kb4 + offA1 + q];
            a[8 + q * 4 + 0] = (_Float16)(x.x - y.x);
            a[8 + q * 4 + 1] = (_Float16)(x.y - y.y);
            a[8 + q * 4 + 2] = (_Float16)(x.z - y.z);
            a[8 + q * 4 + 3] = (_Float16)(x.w - y.w);
        }
        // B (32x16 f16) = A^T: lane<16 holds K 0..7 & 16..23 of col N=lane,
        //                      lane>=16 holds K 8..15 & 24..31 of col N=lane-16.
        #pragma unroll
        for (int q = 0; q < 4; ++q) {
            float4 x = F[sb + kb4 + offB + (q & 1) + 4 * (q >> 1)];
            float4 y = F[db + kb4 + offB + (q & 1) + 4 * (q >> 1)];
            b[q * 4 + 0] = (_Float16)(x.x - y.x);
            b[q * 4 + 1] = (_Float16)(x.y - y.y);
            b[q * 4 + 2] = (_Float16)(x.z - y.z);
            b[q * 4 + 3] = (_Float16)(x.w - y.w);
        }
        acc = __builtin_amdgcn_wmma_f32_16x16x32_f16(false, a, false, b,
                                                     (short)0, acc, false, false);
    }

    // Diagonal of C (16x16 f32): element (m,m) is VGPR m @ lane m for m<8,
    // VGPR m-8 @ lane m+16 for m>=8.
    int idx = -1, eo = -1;
    if (lane < 8)        { idx = lane;      eo = tile * 16 + lane;        }
    else if (lane >= 24) { idx = lane - 24; eo = tile * 16 + (lane - 16); }
    if (idx >= 0) {
        float l2sq = 0.f;
        #pragma unroll
        for (int r = 0; r < 8; ++r) l2sq += (idx == r) ? acc[r] : 0.f;
        float l2 = sqrtf(l2sq);
        att[eo] = 1.0f / (2.0f * l2 + EPS_C);
    }
}

// ---------------------------------------------------------------------------
// Utility / selection kernels
// ---------------------------------------------------------------------------
__global__ void fill_zero_u32(uint32_t* p, size_t n)
{
    size_t i = (size_t)blockIdx.x * blockDim.x + threadIdx.x;
    if (i < n) p[i] = 0u;
}

__global__ void init_sel(uint32_t* sel)
{
    sel[0] = 0u;                 // known high bits (prefix)
    sel[1] = (uint32_t)KDROP;    // remaining k
    sel[2] = 0u;                 // tie ticket counter
}

__global__ void hist_kernel(const float* __restrict__ att,
                            uint32_t* __restrict__ hist,
                            const uint32_t* __restrict__ sel,
                            int shift_known, int shift_field, uint32_t mask)
{
    int i = blockIdx.x * blockDim.x + threadIdx.x;
    if (i >= NE) return;
    uint32_t u = __float_as_uint(att[i]);    // att > 0 => bit order == value order
    if ((u >> shift_known) == sel[0])
        atomicAdd(&hist[(u >> shift_field) & mask], 1u);
}

__global__ void scan_kernel(const uint32_t* __restrict__ hist, uint32_t* sel,
                            int bins, int fieldbits)
{
    uint32_t k = sel[1], known = sel[0], run = 0;
    for (int b = 0; b < bins; ++b) {
        uint32_t c = hist[b];
        if (run + c >= k) {
            sel[0] = (known << fieldbits) | (uint32_t)b;
            sel[1] = k - run;
            return;
        }
        run += c;
    }
}

__global__ void drop_kernel(float* att, uint32_t* sel)
{
    int i = blockIdx.x * blockDim.x + threadIdx.x;
    if (i >= NE) return;
    uint32_t T = sel[0];
    uint32_t u = __float_as_uint(att[i]);
    if (u < T) att[i] = 0.f;
    else if (u == T) {
        if (atomicAdd(&sel[2], 1u) < sel[1]) att[i] = 0.f;  // exact-count ties
    }
}

// ---------------------------------------------------------------------------
// Degree / norm / propagation kernels
// ---------------------------------------------------------------------------
__global__ void degs_kernel(const float* __restrict__ att,
                            const int* __restrict__ dst,
                            float* __restrict__ degs)
{
    int e = blockIdx.x * blockDim.x + threadIdx.x;
    if (e >= NE) return;
    float a = att[e];
    if (a != 0.f) atomicAdd(&degs[dst[e]], a);
}

__global__ void norm_kernel(float* nrm)
{
    int v = blockIdx.x * blockDim.x + threadIdx.x;
    if (v >= NN) return;
    nrm[v] = rsqrtf(nrm[v] + 1.0f);
}

// m = (h*norm)[src] * att  scattered into hacc[dst]
__global__ __launch_bounds__(256) void scatter_kernel(const float* __restrict__ h_in,
                                                      const int* __restrict__ src,
                                                      const int* __restrict__ dst,
                                                      const float* __restrict__ att,
                                                      const float* __restrict__ nrm,
                                                      float* __restrict__ hacc)
{
    const int e = blockIdx.x * 2 + (threadIdx.x >> 7);
    const int k = threadIdx.x & 127;
    float a = att[e];
    if (a == 0.f) return;                     // dropped edges contribute nothing
    int s = src[e], d = dst[e];
    float val = h_in[(size_t)s * DF + k] * (a * nrm[s]);
    atomicAdd(&hacc[(size_t)d * DF + k], val);
}

// h = ALPHA*(hacc*norm) + ALPHA*initial*norm^2 + (1-ALPHA)*h_pre   (in place)
__global__ void combine_kernel(float* __restrict__ h,
                               const float* __restrict__ h_pre,
                               const float* __restrict__ initial,
                               const float* __restrict__ nrm)
{
    size_t i = (size_t)blockIdx.x * blockDim.x + threadIdx.x;
    if (i >= (size_t)NN * DF) return;
    int v = (int)(i >> 7);
    float nv  = nrm[v];
    float out = ALPHA_C * (h[i] * nv) + ALPHA_C * initial[i] * nv * nv
              + (1.0f - ALPHA_C) * h_pre[i];
    h[i] = out;
}

// ---------------------------------------------------------------------------
extern "C" void kernel_launch(void* const* d_in, const int* in_sizes, int n_in,
                              void* d_out, int out_size, void* d_ws, size_t ws_size,
                              hipStream_t stream)
{
    (void)in_sizes; (void)n_in; (void)out_size; (void)ws_size;
    const float* features = (const float*)d_in[0];
    const float* initial  = (const float*)d_in[1];
    const int*   src      = (const int*)d_in[2];
    const int*   dst      = (const int*)d_in[3];
    float*       out      = (float*)d_out;
    float*       wsf      = (float*)d_ws;
    uint32_t*    wsu      = (uint32_t*)d_ws;

    // ws layout (in 4-byte elements), ~28.4 MB total
    const size_t OFF_ATT   = 0;        // 640000 f32
    const size_t OFF_NORM  = 640000;   // 50000 f32 (degs, then norm in place)
    const size_t OFF_HIST1 = 690048;   // 2048 u32
    const size_t OFF_HIST2 = 692096;   // 8192 u32
    const size_t OFF_HIST3 = 700288;   // 256 u32
    const size_t OFF_SEL   = 700544;   // 3 u32
    const size_t OFF_H1    = 700608;   // 6,400,000 f32

    float*    att = wsf + OFF_ATT;
    float*    nrm = wsf + OFF_NORM;
    uint32_t* hg1 = wsu + OFF_HIST1;
    uint32_t* hg2 = wsu + OFF_HIST2;
    uint32_t* hg3 = wsu + OFF_HIST3;
    uint32_t* sel = wsu + OFF_SEL;
    float*    h1  = wsf + OFF_H1;

    // 1) per-edge attention via WMMA Gram diagonals
    att_kernel<<<NE / 16 / 8, 256, 0, stream>>>(features, src, dst, att);

    // 2) zero degs + histograms + sel, init selection state
    {
        size_t n = OFF_H1 - OFF_NORM;
        fill_zero_u32<<<(unsigned)((n + 255) / 256), 256, 0, stream>>>(wsu + OFF_NORM, n);
    }
    init_sel<<<1, 1, 0, stream>>>(sel);

    // 3) exact bottom-KDROP selection: 3-pass radix select (11+13+8 bits)
    hist_kernel<<<NE / 256, 256, 0, stream>>>(att, hg1, sel, 31, 21, 0x7FFu);
    scan_kernel<<<1, 1, 0, stream>>>(hg1, sel, 2048, 11);
    hist_kernel<<<NE / 256, 256, 0, stream>>>(att, hg2, sel, 21, 8, 0x1FFFu);
    scan_kernel<<<1, 1, 0, stream>>>(hg2, sel, 8192, 13);
    hist_kernel<<<NE / 256, 256, 0, stream>>>(att, hg3, sel, 8, 0, 0xFFu);
    scan_kernel<<<1, 1, 0, stream>>>(hg3, sel, 256, 8);
    drop_kernel<<<NE / 256, 256, 0, stream>>>(att, sel);

    // 4) degrees -> norm = (deg+1)^-1/2
    degs_kernel<<<NE / 256, 256, 0, stream>>>(att, dst, nrm);
    norm_kernel<<<(NN + 255) / 256, 256, 0, stream>>>(nrm);

    // 5) iteration 1: initial -> h1
    fill_zero_u32<<<((size_t)NN * DF + 255) / 256, 256, 0, stream>>>((uint32_t*)h1, (size_t)NN * DF);
    scatter_kernel<<<NE / 2, 256, 0, stream>>>(initial, src, dst, att, nrm, h1);
    combine_kernel<<<((size_t)NN * DF + 255) / 256, 256, 0, stream>>>(h1, initial, initial, nrm);

    // 6) iteration 2: h1 -> out
    fill_zero_u32<<<((size_t)NN * DF + 255) / 256, 256, 0, stream>>>((uint32_t*)out, (size_t)NN * DF);
    scatter_kernel<<<NE / 2, 256, 0, stream>>>(h1, src, dst, att, nrm, out);
    combine_kernel<<<((size_t)NN * DF + 255) / 256, 256, 0, stream>>>(out, h1, initial, nrm);
}
